// ConvNorm_86552180949427
// MI455X (gfx1250) — compile-verified, matrix operand order checked by reading, not run on hardware
//
#include <hip/hip_runtime.h>
#include <hip/hip_bf16.h>
#include <cstdint>

typedef __attribute__((ext_vector_type(16))) __bf16 bf16x16;
typedef __attribute__((ext_vector_type(8)))  float  f32x8;
typedef __attribute__((ext_vector_type(4)))  unsigned u32x4;
typedef __attribute__((ext_vector_type(8)))  unsigned u32x8;

#define CIN   256
#define COUT  256
#define HW    4096
#define IMW   64
#define KTOT  2304      // 9 * 256, k = tap*256 + ci
#define NB    4
#define LDT   40        // LDS row stride in bf16 elems (80B: 16B aligned, conflict-free)

__device__ __forceinline__ unsigned short f32_to_bf16(float f) {
  unsigned int u = __float_as_uint(f);
  u += 0x7FFFu + ((u >> 16) & 1u);          // round-to-nearest-even
  return (unsigned short)(u >> 16);
}

union FragU { bf16x16 v; float4 f4[2]; };

__device__ __forceinline__ bf16x16 load_frag(const unsigned short* tile,
                                             int rowBase, int lane) {
  // 16-bit A-operand layout: lane = row (mod 16); lane>>4 picks K chunks
  // {0..7,16..23} vs {8..15,24..31}; two 16B contiguous LDS reads.
  int row = rowBase + (lane & 15);
  int kh  = (lane >> 4) & 1;
  const unsigned short* p = tile + row * LDT + kh * 8;
  FragU f;
  f.f4[0] = *reinterpret_cast<const float4*>(p);
  f.f4[1] = *reinterpret_cast<const float4*>(p + 16);
  return f.v;
}

__device__ __forceinline__ unsigned lds_addr_of(void* p) {
  return (unsigned)(unsigned long long)(__attribute__((address_space(3))) char*)p;
}

__device__ __forceinline__ void wait_tensor0() {
#if __has_builtin(__builtin_amdgcn_s_wait_tensorcnt)
  __builtin_amdgcn_s_wait_tensorcnt(0);
#else
  asm volatile("s_wait_tensorcnt 0x0" ::: "memory");
#endif
}

// TDM: DMA a 256-row x 64B tile (row stride 4608B) from global into LDS,
// padding 16B after every 64B row -> 80B LDS row stride (== LDT).
// D# group0: count=1 | lds_addr | global_addr[56:0] | type=2
// D# group1: data_size=4B, pad_enable, pad_interval=16DW, pad_amount=4DW,
//            tensor_dim0=1152 (4B units), tensor_dim1=256, tile_dim0=16,
//            tile_dim1=256, tensor_dim0_stride=1152
__device__ __forceinline__ void tdm_load_a(const unsigned short* gsrc,
                                           unsigned ldsOff) {
  unsigned long long ga = (unsigned long long)gsrc;
  u32x4 g0;
  g0[0] = 1u;                                            // count=1, user mode
  g0[1] = ldsOff;                                        // lds_addr
  g0[2] = (unsigned)ga;                                  // global_addr[31:0]
  g0[3] = (unsigned)((ga >> 32) & 0x1FFFFFFull) | (2u << 30); // [56:32] | type=2
  u32x8 g1;
  g1[0] = (2u << 16) | (1u << 20) | (3u << 22) | (3u << 25); // 4B | pad_en | 16DW | 4DW
  g1[1] = (1152u & 0xFFFFu) << 16;                       // tensor_dim0 lo16
  g1[2] = (1152u >> 16) | (256u << 16);                  // dim0 hi16 | tensor_dim1 lo16
  g1[3] = (256u >> 16) | (16u << 16);                    // dim1 hi16 | tile_dim0=16
  g1[4] = 256u;                                          // tile_dim1=256, tile_dim2=0
  g1[5] = 1152u;                                         // tensor_dim0_stride lo32
  g1[6] = 0u;                                            // stride hi | dim1_stride lo
  g1[7] = 0u;
  asm volatile("tensor_load_to_lds %0, %1" :: "s"(g0), "s"(g1) : "memory");
}

// ---- Kernel 1: pad+permute offset-conv weight to bf16 A[32][KTOT], k=tap*256+ci ----
__global__ void __launch_bounds__(256)
woffcvt_kernel(const float* __restrict__ w_off, unsigned short* __restrict__ A) {
  int tid = blockIdx.x * 256 + threadIdx.x;           // co * KTOT + k, co < 32
  int k   = tid % KTOT;
  int co  = tid / KTOT;
  int tap = k >> 8, ci = k & 255;
  float w = (co < 27) ? w_off[((size_t)co * CIN + ci) * 9 + tap] : 0.f;
  A[tid] = f32_to_bf16(w);
}

// ---- Kernel 2: offset/mask conv as WMMA GEMM: M=32(27), N=64/block, K=2304 ----
__global__ void __launch_bounds__(256)
offconv_wmma_kernel(const float* __restrict__ x, const unsigned short* __restrict__ A,
                    const float* __restrict__ b_off, float* __restrict__ om) {
  __shared__ __align__(16) unsigned short Atile[32 * LDT];
  __shared__ __align__(16) unsigned short Btile[64 * LDT];

  const int tid  = threadIdx.x;
  const int lane = tid & 31;
  const int wv   = tid >> 5;
  const int n0   = blockIdx.x * 64;          // flat (b,pixel) tile
  const int b    = n0 >> 12;
  const int pb   = n0 & 4095;

  const int am = tid >> 2;                   // A-staging row (only <32 used)
  const int ak = (tid & 3) * 8;

  const int bn  = tid >> 2;                  // B-build pixel-in-tile
  const int bk  = (tid & 3) * 8;             // channel chunk
  const int p   = pb + bn;
  const int oh  = p >> 6, ow = p & 63;

  f32x8 acc = {0.f, 0.f, 0.f, 0.f, 0.f, 0.f, 0.f, 0.f};
  const int mFrag = (wv & 1) * 16;
  const int nFrag = (wv >> 1) * 16;

  for (int ks = 0; ks < KTOT / 32; ++ks) {
    const int kbase = ks * 32;
    const int tap   = kbase >> 8;
    const int ci    = (kbase & 255) + bk;

    if (am < 32)
      *reinterpret_cast<float4*>(&Atile[am * LDT + ak]) =
          *reinterpret_cast<const float4*>(A + (size_t)am * KTOT + kbase + ak);

    {
      int y  = oh - 1 + tap / 3;
      int xx = ow - 1 + tap % 3;
      bool valid = ((unsigned)y < 64u) && ((unsigned)xx < 64u);
      int idx = y * IMW + xx;
      const float* xp = x + (size_t)(b * CIN + ci) * HW;
#pragma unroll
      for (int j = 0; j < 8; ++j) {
        float v = valid ? xp[idx] : 0.f;
        Btile[bn * LDT + bk + j] = f32_to_bf16(v);
        xp += HW;
      }
    }
    __syncthreads();
    bf16x16 af = load_frag(Atile, mFrag, lane);
    bf16x16 bf = load_frag(Btile, nFrag, lane);
    acc = __builtin_amdgcn_wmma_f32_16x16x32_bf16(false, af, false, bf,
                                                  (short)0, acc, false, false);
    __syncthreads();
  }

  const int nlane = lane & 15;
  const int moff  = (lane >> 4) * 8;
  const int col   = n0 + nFrag + nlane;
  const int pcol  = col & 4095;
#pragma unroll
  for (int r = 0; r < 8; ++r) {
    int co = mFrag + moff + r;
    if (co < 27)
      om[((size_t)(b * 27 + co) << 12) + pcol] = acc[r] + b_off[co];
  }
}

// ------- Kernel 3: per (b,tap,pixel) bilinear corner weights(+mask) & indices -------
__global__ void __launch_bounds__(256)
coords_kernel(const float* __restrict__ om, float4* __restrict__ cw,
              int4* __restrict__ cidx) {
  int tid = blockIdx.x * 256 + threadIdx.x;           // (b, tap, p)
  int p   = tid & (HW - 1);
  int tap = (tid >> 12) % 9;
  int b   = tid / (9 * HW);
  const float* omb = om + (size_t)b * 27 * HW;
  float offy = omb[tap * HW + p];
  float offx = omb[(9 + tap) * HW + p];
  float mask = 1.0f / (1.0f + __expf(-omb[(18 + tap) * HW + p]));
  int oh = p >> 6, ow = p & 63;
  float py = (float)(oh - 1 + tap / 3) + offy;
  float px = (float)(ow - 1 + tap % 3) + offx;
  float fy = floorf(py), fx = floorf(px);
  float wy1 = py - fy, wx1 = px - fx;
  int y0 = (int)fy, x0 = (int)fx, y1 = y0 + 1, x1 = x0 + 1;
  float vy0 = (y0 >= 0 && y0 < 64) ? 1.f : 0.f;
  float vy1 = (y1 >= 0 && y1 < 64) ? 1.f : 0.f;
  float vx0 = (x0 >= 0 && x0 < 64) ? 1.f : 0.f;
  float vx1 = (x1 >= 0 && x1 < 64) ? 1.f : 0.f;
  float w00 = (1.f - wy1) * (1.f - wx1) * vy0 * vx0 * mask;
  float w01 = (1.f - wy1) * wx1        * vy0 * vx1 * mask;
  float w10 = wy1 * (1.f - wx1)        * vy1 * vx0 * mask;
  float w11 = wy1 * wx1                * vy1 * vx1 * mask;
  int yc0 = min(max(y0, 0), 63), yc1 = min(max(y1, 0), 63);
  int xc0 = min(max(x0, 0), 63), xc1 = min(max(x1, 0), 63);
  cw[tid]   = make_float4(w00, w01, w10, w11);
  cidx[tid] = make_int4(yc0 * IMW + xc0, yc0 * IMW + xc1,
                        yc1 * IMW + xc0, yc1 * IMW + xc1);
}

// ---- Kernel 4: main weight fp32 [co][ci][tap] -> bf16 A[co][tap*256+ci] ----
__global__ void __launch_bounds__(256)
wcvt_kernel(const float* __restrict__ weight, unsigned short* __restrict__ A) {
  int tid = blockIdx.x * 256 + threadIdx.x;           // co * KTOT + k
  int k   = tid % KTOT;
  int co  = tid / KTOT;
  int tap = k >> 8, ci = k & 255;
  A[tid] = f32_to_bf16(weight[((size_t)co * CIN + ci) * 9 + tap]);
}

// ---- Kernel 5: fused sample + bf16 WMMA GEMM, full M=256 per block ----
// Block tile M=256 x N=64; A tile streamed by the Tensor Data Mover into
// double-buffered LDS (one tensor_load_to_lds per K-step, overlapped with
// WMMA); B tile built by fused bilinear gather. One barrier per K-step.
__global__ void __launch_bounds__(256)
dcn_gemm_kernel(const float* __restrict__ x, const unsigned short* __restrict__ A,
                const float4* __restrict__ cw, const int4* __restrict__ cidx,
                const float* __restrict__ bias, float* __restrict__ out) {
  __shared__ __align__(16) unsigned short Atile[2][256 * LDT];
  __shared__ __align__(16) unsigned short Btile[2][64 * LDT];

  const int tid  = threadIdx.x;
  const int lane = tid & 31;
  const int wv   = tid >> 5;                 // 8 waves
  const int n0   = blockIdx.x * 64;          // flat (b,pixel) tile
  const int b    = n0 >> 12;
  const int pb   = n0 & 4095;

  const int bn   = tid >> 2;                 // B build pixel-in-tile
  const int bk   = (tid & 3) * 8;
  const int pcol = pb + bn;

  const unsigned ldsA0 = lds_addr_of(&Atile[0][0]);
  const unsigned ldsA1 = lds_addr_of(&Atile[1][0]);

  f32x8 acc[2][4];
#pragma unroll
  for (int i = 0; i < 2; ++i)
#pragma unroll
    for (int j = 0; j < 4; ++j)
      acc[i][j] = f32x8{0.f, 0.f, 0.f, 0.f, 0.f, 0.f, 0.f, 0.f};

  const int mBase = wv * 32;                 // wave's 32-row M strip
  const int NS = KTOT / 32;                  // 72 K-steps

  if (wv == 0) tdm_load_a(A, ldsA0);         // prologue DMA for ks=0

  for (int ks = 0; ks < NS; ++ks) {
    const int cur   = ks & 1;
    const int kbase = ks * 32;
    const int tap   = kbase >> 8;            // K step stays within one tap
    const int ci    = (kbase & 255) + bk;

    // build B tile via fused bilinear gather (x resident in 192MB L2)
    {
      const int cq   = (b * 9 + tap) * HW + pcol;
      const float4 w = cw[cq];
      const int4  id = cidx[cq];
      const float* xp = x + (size_t)(b * CIN + ci) * HW;
#pragma unroll
      for (int j = 0; j < 8; ++j) {
        float v = w.x * xp[id.x] + w.y * xp[id.y]
                + w.z * xp[id.z] + w.w * xp[id.w];
        Btile[cur][bn * LDT + bk + j] = f32_to_bf16(v);
        xp += HW;
      }
    }
    if (tap < 8)                                   // prefetch next tap's coords
      __builtin_prefetch(cw + (b * 9 + tap + 1) * HW + pcol, 0, 3);

    if (wv == 0) wait_tensor0();             // current A tile landed in LDS
    __syncthreads();
    // issue next DMA only after the barrier: all reads of the other buffer
    // (iteration ks-1) are complete by now.
    if (wv == 0 && ks + 1 < NS)
      tdm_load_a(A + (size_t)(ks + 1) * 32, cur ? ldsA0 : ldsA1);

    bf16x16 a0 = load_frag(Atile[cur], mBase, lane);
    bf16x16 a1 = load_frag(Atile[cur], mBase + 16, lane);
#pragma unroll
    for (int j = 0; j < 4; ++j) {
      bf16x16 bf = load_frag(Btile[cur], j * 16, lane);
      acc[0][j] = __builtin_amdgcn_wmma_f32_16x16x32_bf16(false, a0, false, bf,
                                                          (short)0, acc[0][j],
                                                          false, false);
      acc[1][j] = __builtin_amdgcn_wmma_f32_16x16x32_bf16(false, a1, false, bf,
                                                          (short)0, acc[1][j],
                                                          false, false);
    }
  }

  // epilogue: C/D layout — VGPR r holds M=r (lanes 0-15) / M=r+8 (lanes 16-31)
  const int nlane = lane & 15;
  const int moff  = (lane >> 4) * 8;
#pragma unroll
  for (int i = 0; i < 2; ++i) {
#pragma unroll
    for (int r = 0; r < 8; ++r) {
      int co = mBase + i * 16 + moff + r;
      float bv = bias[co];
      size_t base = (size_t)(b * COUT + co) * HW;
#pragma unroll
      for (int j = 0; j < 4; ++j) {
        int p = (n0 + j * 16 + nlane) & 4095;
        out[base + p] = acc[i][j][r] + bv;
      }
    }
  }
}

// ---------------- Kernel 6/7: GroupNorm ----------------
__global__ void __launch_bounds__(256)
gn_stats_kernel(const float* __restrict__ out, float2* __restrict__ stats) {
  int bg = blockIdx.x;                       // (b, g)
  int b = bg >> 4, g = bg & 15;
  const float* base = out + (size_t)(b * COUT + g * 16) * HW;
  float s = 0.f, ss = 0.f;
  for (int i = threadIdx.x; i < 16 * HW; i += 256) {
    float v = base[i];
    s += v; ss += v * v;
  }
  __shared__ float rs[256], rss[256];
  rs[threadIdx.x] = s; rss[threadIdx.x] = ss;
  __syncthreads();
  for (int st = 128; st > 0; st >>= 1) {
    if (threadIdx.x < st) {
      rs[threadIdx.x]  += rs[threadIdx.x + st];
      rss[threadIdx.x] += rss[threadIdx.x + st];
    }
    __syncthreads();
  }
  if (threadIdx.x == 0) {
    float inv = 1.0f / (float)(16 * HW);
    float mu  = rs[0] * inv;
    float var = rss[0] * inv - mu * mu;
    stats[bg] = make_float2(mu, rsqrtf(var + 1e-5f));
  }
}

__global__ void __launch_bounds__(256)
gn_apply_kernel(float* __restrict__ out, const float2* __restrict__ stats,
                const float* __restrict__ gamma, const float* __restrict__ beta) {
  size_t tid = (size_t)blockIdx.x * 256 + threadIdx.x;  // b*2^20 + co*2^12 + p
  int co = (int)((tid >> 12) & 255);
  int b  = (int)(tid >> 20);
  float2 st = stats[b * 16 + (co >> 4)];
  float v = out[tid];
  out[tid] = (v - st.x) * st.y * gamma[co] + beta[co];
}

// ---------------- launch ----------------
extern "C" void kernel_launch(void* const* d_in, const int* in_sizes, int n_in,
                              void* d_out, int out_size, void* d_ws, size_t ws_size,
                              hipStream_t stream) {
  const float* x      = (const float*)d_in[0];
  const float* w_off  = (const float*)d_in[1];
  const float* b_off  = (const float*)d_in[2];
  const float* weight = (const float*)d_in[3];
  const float* bias   = (const float*)d_in[4];
  const float* gamma  = (const float*)d_in[5];
  const float* beta   = (const float*)d_in[6];
  float* out = (float*)d_out;

  char* ws = (char*)d_ws;
  float*          om    = (float*)(ws);                         // 1,769,472 B
  float4*         cw    = (float4*)(ws + 1769472);              // 2,359,296 B
  int4*           cidx  = (int4*)(ws + 4128768);                // 2,359,296 B
  unsigned short* Abf   = (unsigned short*)(ws + 6488064);      // 1,179,648 B
  unsigned short* Aoff  = (unsigned short*)(ws + 7667712);      //   147,456 B
  float2*         stats = (float2*)(ws + 7815168);

  woffcvt_kernel<<<(32 * KTOT) / 256, 256, 0, stream>>>(w_off, Aoff);
  wcvt_kernel<<<(COUT * KTOT) / 256, 256, 0, stream>>>(weight, Abf);
  offconv_wmma_kernel<<<(NB * HW) / 64, 256, 0, stream>>>(x, Aoff, b_off, om);
  coords_kernel<<<(NB * 9 * HW) / 256, 256, 0, stream>>>(om, cw, cidx);
  dcn_gemm_kernel<<<(NB * HW) / 64, 256, 0, stream>>>(x, Abf, cw, cidx, bias, out);
  gn_stats_kernel<<<NB * 16, 256, 0, stream>>>(out, stats);
  gn_apply_kernel<<<(unsigned)(out_size / 256), 256, 0, stream>>>(out, stats, gamma, beta);
}
